// SPINN_56487409877566
// MI455X (gfx1250) — compile-verified, matrix operand order checked by reading, not run on hardware
//
#include <hip/hip_runtime.h>
#include <math.h>

typedef __attribute__((ext_vector_type(16))) _Float16 v16h;
typedef __attribute__((ext_vector_type(8)))  _Float16 v8h;
typedef __attribute__((ext_vector_type(8)))  float    v8f;
typedef __attribute__((ext_vector_type(4)))  float    v4f;

#define DD 128
#define BB 64
#define LL 96
#define TSTEPS 189
#define SS (LL + 2)          // stack slots per example
#define ASTR 392             // Abuf row stride in halves (384 + 8 pad -> bank-friendly)
#define THSTR 136            // Th row stride in halves (128 + 8 pad)

// workspace byte offsets
#define WIH_OFF   0u         // 512*384 f16
#define WHH_OFF   393216u    // 512*128 f16
#define WCAT_OFF  524288u    // 640*384 f16
#define STACK_OFF 1015808u   // 64*98*256 f32

// branchless fast sigmoid/tanh on v_exp_f32 (no libm branches on the critical path)
__device__ __forceinline__ float sigm(float x) {
  return __fdividef(1.0f, 1.0f + __expf(-x));
}
__device__ __forceinline__ float tanh_fast(float x) {
  // 1 - 2/(e^{2x}+1); e^{2x}->inf gives 1, e^{2x}->0 gives -1 (branchless)
  return 1.0f - __fdividef(2.0f, __expf(2.0f * x) + 1.0f);
}

// Load one 16x32 (A) or 32x16-as-B fragment row slice with the CDNA5 per-lane K swizzle:
// lanes 0-15: K = k0+{0..7}, k0+{16..23}; lanes 16-31: K = k0+{8..15}, k0+{24..31}.
__device__ __forceinline__ v16h load_frag(const _Float16* rowbase, int k0, int lane) {
  const _Float16* p = rowbase + k0 + ((lane & 16) ? 8 : 0);
  v8h lo = *(const v8h*)(p);
  v8h hi = *(const v8h*)(p + 16);
  v16h r;
#pragma unroll
  for (int i = 0; i < 8; ++i) { r[i] = lo[i]; r[i + 8] = hi[i]; }
  return r;
}

// One-shot: cast weights to f16 in workspace; permute W_ih columns so the
// A layout [s2h | s1h | slot3] serves both GEMMs. Wcat = [Wl | Wr | Wt] per row.
__global__ void spinn_convert(const float* __restrict__ W_ih,
                              const float* __restrict__ W_hh,
                              const float* __restrict__ Wl,
                              const float* __restrict__ Wr,
                              const float* __restrict__ Wt,
                              _Float16* __restrict__ Wih_h,
                              _Float16* __restrict__ Whh_h,
                              _Float16* __restrict__ Wcat_h) {
  int i = blockIdx.x * blockDim.x + threadIdx.x;
  if (i < 512 * 384) {
    int n = i / 384, k = i - n * 384;
    int ks = (k < 128) ? (k + 256) : ((k < 256) ? k : (k - 256));
    Wih_h[i] = (_Float16)W_ih[n * 384 + ks];
  }
  if (i < 512 * 128) {
    Whh_h[i] = (_Float16)W_hh[i];
  }
  if (i < 640 * 384) {
    int n = i / 384, k = i - n * 384;
    float v = (k < 128) ? Wl[n * 128 + k]
            : (k < 256) ? Wr[n * 128 + (k - 128)]
                        : Wt[n * 128 + (k - 256)];
    Wcat_h[i] = (_Float16)v;
  }
}

__global__ __launch_bounds__(512, 1)
void spinn_scan(const float* __restrict__ buffers,
                const int* __restrict__ trans,
                const float* __restrict__ b_ih,
                const float* __restrict__ b_hh,
                const float* __restrict__ bl,
                const _Float16* __restrict__ Wih_h,
                const _Float16* __restrict__ Whh_h,
                const _Float16* __restrict__ Wcat_h,
                float* __restrict__ stackg,
                float* __restrict__ out) {
  __shared__ _Float16 Abuf[BB * ASTR];   // [64 x 384(+pad)] f16 activations
  __shared__ _Float16 Th[BB * THSTR];    // tracker h, f16
  __shared__ int sp_s[BB];
  __shared__ int bp_s[BB];

  const int tid  = threadIdx.x;
  const int lane = tid & 31;
  const int wave = tid >> 5;
  const int ln15 = lane & 15;
  const int hsel = (lane >> 4) & 1;

  // Partition: wave owns column group ng = wave&7 (16 cols of every 128-wide gate)
  // and BOTH m-tiles mt = 2*mh + g (mh = wave>>3, g in {0,1}).
  // B (weight) fragments are loaded once per kk and reused for both m-tiles.
  const int ng = wave & 7;
  const int mh = wave >> 3;
  const int ncol = ng * 16 + ln15;       // column within each 128-wide gate

  // tracker cell state, register-resident per (g, r): b = (2*mh+g)*16 + r + 8*hsel
  float tcs[2][8];
#pragma unroll
  for (int g = 0; g < 2; ++g)
#pragma unroll
    for (int r = 0; r < 8; ++r) tcs[g][r] = 0.0f;

  for (int i = tid; i < BB * THSTR; i += 512) Th[i] = (_Float16)0.0f;
  if (tid < BB) { sp_s[tid] = 2; bp_s[tid] = LL; }
  // stack slots 0,1 <- buffers[0][b]
  for (int i = tid; i < BB * 2 * 256; i += 512) {
    int b = i >> 9;
    int rem = i & 511;
    int slot = rem >> 8;
    int c = rem & 255;
    stackg[(b * SS + slot) * 256 + c] = buffers[b * 256 + c];
  }
  __threadfence();
  __syncthreads();

#pragma unroll 1
  for (int t = 0; t < TSTEPS; ++t) {
    // ---- phase 1: build A = [s2h | s1h | buf_top_h] as f16 in LDS ----
#pragma unroll 1
    for (int i = tid; i < BB * 48; i += 512) {
      int b = i / 48;
      int col = (i - b * 48) * 8;
      int sp = sp_s[b], bp = bp_s[b];
      const float* src;
      if (col < 128)      src = stackg + (b * SS + sp - 2) * 256 + col;
      else if (col < 256) src = stackg + (b * SS + sp - 1) * 256 + (col - 128);
      else                src = buffers + ((bp - 1) * BB + b) * 256 + (col - 256);
      v4f x0 = *(const v4f*)src;
      v4f x1 = *(const v4f*)(src + 4);
      v8h h;
#pragma unroll
      for (int j = 0; j < 4; ++j) { h[j] = (_Float16)x0[j]; h[j + 4] = (_Float16)x1[j]; }
      *(v8h*)(Abuf + b * ASTR + col) = h;
    }
    __syncthreads();

    // ---- phase 2: tracker GEMM (WMMA, B reused across both m-tiles) + LSTM cell ----
    float th2v[2][8];
    {
      v8f acc[2][4];                     // [m-tile g][gate q]
#pragma unroll
      for (int g = 0; g < 2; ++g)
#pragma unroll
        for (int q = 0; q < 4; ++q)
#pragma unroll
          for (int r = 0; r < 8; ++r) acc[g][q][r] = 0.0f;

      const _Float16* Arow0 = Abuf + ((mh * 2 + 0) * 16 + ln15) * ASTR;
      const _Float16* Arow1 = Abuf + ((mh * 2 + 1) * 16 + ln15) * ASTR;
#pragma unroll 1
      for (int kk = 0; kk < 12; ++kk) {          // x @ W_ih'
        v16h a0 = load_frag(Arow0, kk * 32, lane);
        v16h a1 = load_frag(Arow1, kk * 32, lane);
#pragma unroll
        for (int q = 0; q < 4; ++q) {
          v16h bf = load_frag(Wih_h + (q * 128 + ncol) * 384, kk * 32, lane);
          acc[0][q] = __builtin_amdgcn_wmma_f32_16x16x32_f16(false, a0, false, bf,
                                                             (short)0, acc[0][q], false, false);
          acc[1][q] = __builtin_amdgcn_wmma_f32_16x16x32_f16(false, a1, false, bf,
                                                             (short)0, acc[1][q], false, false);
        }
      }
      const _Float16* Trow0 = Th + ((mh * 2 + 0) * 16 + ln15) * THSTR;
      const _Float16* Trow1 = Th + ((mh * 2 + 1) * 16 + ln15) * THSTR;
#pragma unroll 1
      for (int kk = 0; kk < 4; ++kk) {           // th @ W_hh
        v16h a0 = load_frag(Trow0, kk * 32, lane);
        v16h a1 = load_frag(Trow1, kk * 32, lane);
#pragma unroll
        for (int q = 0; q < 4; ++q) {
          v16h bf = load_frag(Whh_h + (q * 128 + ncol) * 128, kk * 32, lane);
          acc[0][q] = __builtin_amdgcn_wmma_f32_16x16x32_f16(false, a0, false, bf,
                                                             (short)0, acc[0][q], false, false);
          acc[1][q] = __builtin_amdgcn_wmma_f32_16x16x32_f16(false, a1, false, bf,
                                                             (short)0, acc[1][q], false, false);
        }
      }
      float bi = b_ih[ncol]       + b_hh[ncol];
      float bf = b_ih[128 + ncol] + b_hh[128 + ncol];
      float bg = b_ih[256 + ncol] + b_hh[256 + ncol];
      float bo = b_ih[384 + ncol] + b_hh[384 + ncol];
#pragma unroll
      for (int g = 0; g < 2; ++g)
#pragma unroll
        for (int r = 0; r < 8; ++r) {
          float c2 = sigm(acc[g][1][r] + bf) * tcs[g][r]
                   + sigm(acc[g][0][r] + bi) * tanh_fast(acc[g][2][r] + bg);
          tcs[g][r] = c2;
          th2v[g][r] = sigm(acc[g][3][r] + bo) * tanh_fast(c2);
        }
    }
    __syncthreads();   // everyone done reading old Th / Abuf slot3

    // publish th2 into Th (next step) and Abuf[:,256:384] (composition A)
#pragma unroll
    for (int g = 0; g < 2; ++g) {
#pragma unroll
      for (int r = 0; r < 8; ++r) {
        int m = (mh * 2 + g) * 16 + r + hsel * 8;
        _Float16 hv = (_Float16)th2v[g][r];
        Th[m * THSTR + ncol] = hv;
        Abuf[m * ASTR + 256 + ncol] = hv;
      }
    }
    __syncthreads();

    // ---- phase 3: composition GEMM (WMMA): [s2h|s1h|th2] @ [Wl|Wr|Wt]^T ----
    v8f cacc[2][5];
    {
#pragma unroll
      for (int g = 0; g < 2; ++g)
#pragma unroll
        for (int q = 0; q < 5; ++q)
#pragma unroll
          for (int r = 0; r < 8; ++r) cacc[g][q][r] = 0.0f;
      const _Float16* Arow0 = Abuf + ((mh * 2 + 0) * 16 + ln15) * ASTR;
      const _Float16* Arow1 = Abuf + ((mh * 2 + 1) * 16 + ln15) * ASTR;
#pragma unroll 1
      for (int kk = 0; kk < 12; ++kk) {
        v16h a0 = load_frag(Arow0, kk * 32, lane);
        v16h a1 = load_frag(Arow1, kk * 32, lane);
#pragma unroll
        for (int q = 0; q < 5; ++q) {
          v16h bf = load_frag(Wcat_h + (q * 128 + ncol) * 384, kk * 32, lane);
          cacc[0][q] = __builtin_amdgcn_wmma_f32_16x16x32_f16(false, a0, false, bf,
                                                              (short)0, cacc[0][q], false, false);
          cacc[1][q] = __builtin_amdgcn_wmma_f32_16x16x32_f16(false, a1, false, bf,
                                                              (short)0, cacc[1][q], false, false);
        }
      }
    }

    // ---- phase 4: composition cell + stack update (reduce RAW is same-thread) ----
    {
      float bg_ = bl[ncol];
      float bi_ = bl[128 + ncol];
      float bf1 = bl[256 + ncol];
      float bf2 = bl[384 + ncol];
      float bo_ = bl[512 + ncol];
#pragma unroll
      for (int g = 0; g < 2; ++g) {
#pragma unroll
        for (int r = 0; r < 8; ++r) {
          int b = (mh * 2 + g) * 16 + r + hsel * 8;   // batch row
          int sp = sp_s[b], bp = bp_s[b];
          int fl = trans[t * BB + b];
          if (fl == 3) {                   // shift: push buf_top
            const float* btop = buffers + ((bp - 1) * BB + b) * 256;
            float* dst = stackg + (b * SS + sp) * 256;
            dst[ncol] = btop[ncol];
            dst[128 + ncol] = btop[128 + ncol];
          } else if (fl == 2) {            // reduce: compose and overwrite slot sp-2
            float s2c = stackg[(b * SS + sp - 2) * 256 + 128 + ncol];
            float s1c = stackg[(b * SS + sp - 1) * 256 + 128 + ncol];
            float g_ = cacc[g][0][r] + bg_;
            float i_ = cacc[g][1][r] + bi_;
            float f1 = cacc[g][2][r] + bf1;
            float f2 = cacc[g][3][r] + bf2;
            float o_ = cacc[g][4][r] + bo_;
            float c  = sigm(f1) * s2c + sigm(f2) * s1c + sigm(i_) * tanh_fast(g_);
            float h  = sigm(o_) * tanh_fast(c);
            float* dst = stackg + (b * SS + sp - 2) * 256;
            dst[ncol] = h;
            dst[128 + ncol] = c;
          }
        }
      }
    }
    __threadfence();
    __syncthreads();
    if (tid < BB) {
      int fl = trans[t * BB + tid];
      if (fl == 3)      { sp_s[tid] += 1; bp_s[tid] -= 1; }
      else if (fl == 2) { sp_s[tid] -= 1; }
    }
    __syncthreads();
  }

  // output: h-part of final stack top, [64,128] f32
  for (int i = tid; i < BB * DD; i += 512) {
    int b = i >> 7;
    int c = i & 127;
    out[i] = stackg[(b * SS + sp_s[b] - 1) * 256 + c];
  }
}

extern "C" void kernel_launch(void* const* d_in, const int* in_sizes, int n_in,
                              void* d_out, int out_size, void* d_ws, size_t ws_size,
                              hipStream_t stream) {
  const float* buffers = (const float*)d_in[0];
  const int*   trans   = (const int*)d_in[1];
  const float* W_ih    = (const float*)d_in[2];
  const float* W_hh    = (const float*)d_in[3];
  const float* b_ih    = (const float*)d_in[4];
  const float* b_hh    = (const float*)d_in[5];
  const float* Wl      = (const float*)d_in[6];
  const float* bl      = (const float*)d_in[7];
  const float* Wr      = (const float*)d_in[8];
  const float* Wt      = (const float*)d_in[9];

  char* ws = (char*)d_ws;
  _Float16* Wih_h  = (_Float16*)(ws + WIH_OFF);
  _Float16* Whh_h  = (_Float16*)(ws + WHH_OFF);
  _Float16* Wcat_h = (_Float16*)(ws + WCAT_OFF);
  float*    stackg = (float*)(ws + STACK_OFF);

  spinn_convert<<<dim3((640 * 384 + 255) / 256), dim3(256), 0, stream>>>(
      W_ih, W_hh, Wl, Wr, Wt, Wih_h, Whh_h, Wcat_h);
  spinn_scan<<<dim3(1), dim3(512), 0, stream>>>(
      buffers, trans, b_ih, b_hh, bl, Wih_h, Whh_h, Wcat_h, stackg, (float*)d_out);
}
